// Decoder_70325794504892
// MI455X (gfx1250) — matrix operand order repaired and checked by reading, not compiled, not run on hardware
//
#include <hip/hip_runtime.h>

// ---------------------------------------------------------------------------
// Problem constants (B=32, T=64, S=64, V=32000, E=512, H=512, L=2)
// ---------------------------------------------------------------------------
#define Bn 32
#define Tn 64
#define Sn 64
#define Vn 32000
#define En 512
#define Hn 512

typedef __attribute__((ext_vector_type(8)))  __bf16 v8bf;
typedef __attribute__((ext_vector_type(16))) __bf16 v16bf;
typedef __attribute__((ext_vector_type(8)))  float  v8f;

__device__ __forceinline__ v16bf cat8(v8bf lo, v8bf hi) {
  return __builtin_shufflevector(lo, hi, 0,1,2,3,4,5,6,7,8,9,10,11,12,13,14,15);
}
__device__ __forceinline__ float sigf(float x) { return 1.f / (1.f + __expf(-x)); }

// ---------------------------------------------------------------------------
// Tensor Data Mover support (6-arg upstream-clang form, confirmed by compile
// probe: (uint32x4 g0, int32x8 g1, int32x4, int32x4, int32x8, i32 cpol))
// ---------------------------------------------------------------------------
#if defined(__has_builtin)
#if __has_builtin(__builtin_amdgcn_tensor_load_to_lds) && \
    __has_builtin(__builtin_amdgcn_s_wait_tensorcnt)
#define USE_TDM 1
#endif
#endif

#if defined(USE_TDM)
typedef unsigned int v4u __attribute__((ext_vector_type(4)));
typedef int          v4i __attribute__((ext_vector_type(4)));
typedef int          v8i __attribute__((ext_vector_type(8)));

// DMA a [rows x 64] bf16 tile (row stride = stride_elems) into LDS at byte
// offset lds_addr, padding +16B after every 128B row (LDS row stride 72 elems).
// D# encoding per CDNA5 ISA 8.3/8.4: count=1, type=2, data_size=1 (2 bytes),
// pad_enable, pad_interval=32 DWORDs (code 4), pad_amount=4 DWORDs (code 3).
__device__ __forceinline__ void tdm_load_tile(const __bf16* gsrc, unsigned lds_addr,
                                              int rows, int stride_elems) {
  unsigned long long ga = (unsigned long long)(size_t)gsrc;
  v4u g0;
  g0.x = 1u;                                                  // count=1, user mode
  g0.y = lds_addr;                                            // lds_addr [63:32]
  g0.z = (unsigned)(ga & 0xFFFFFFFFu);                        // global_addr lo
  g0.w = (unsigned)((ga >> 32) & 0x01FFFFFFu) | (2u << 30);   // global_addr hi | type=2
  unsigned K = (unsigned)stride_elems, R = (unsigned)rows;
  v8i g1;
  g1[0] = (int)((1u << 16) | (1u << 20) | (4u << 22) | (3u << 25)); // data_size=2B, pad cfg
  g1[1] = (int)((K & 0xFFFFu) << 16);                               // tensor_dim0 lo16
  g1[2] = (int)((K >> 16) | ((R & 0xFFFFu) << 16));                 // dim0 hi16 | dim1 lo16
  g1[3] = (int)((R >> 16) | (64u << 16));                           // dim1 hi16 | tile_dim0=64
  g1[4] = (int)(R & 0xFFFFu);                                       // tile_dim1 = rows
  g1[5] = (int)K;                                                   // dim0_stride lo32
  g1[6] = 0;
  g1[7] = 0;
  v4i z4 = (v4i){0, 0, 0, 0};
  v8i z8 = (v8i){0, 0, 0, 0, 0, 0, 0, 0};
  __builtin_amdgcn_tensor_load_to_lds(g0, g1, z4, z4, z8, 0);
}
#endif

// ---------------------------------------------------------------------------
// Elementwise fp32 -> bf16 convert
// ---------------------------------------------------------------------------
__global__ void k_cvt_bf16(const float* __restrict__ in, __bf16* __restrict__ out, int n) {
  int i = blockIdx.x * 256 + threadIdx.x;
  if (i < n) out[i] = (__bf16)in[i];
}

// Fuse [W_ih | W_hh] -> Wcat bf16, rows = 2048, K = kih+khh
__global__ void k_cat_w(const float* __restrict__ wih, const float* __restrict__ whh,
                        __bf16* __restrict__ out, int kih, int khh) {
  int kt = kih + khh;
  int i = blockIdx.x * 256 + threadIdx.x;
  if (i >= 2048 * kt) return;
  int nrow = i / kt, k = i - nrow * kt;
  float v = (k < kih) ? wih[(size_t)nrow * kih + k] : whh[(size_t)nrow * khh + (k - kih)];
  out[i] = (__bf16)v;
}

// 512x512 transpose + convert: out[n,k] = in[k,n]
__global__ void k_transpose512(const float* __restrict__ in, __bf16* __restrict__ out) {
  int i = blockIdx.x * 256 + threadIdx.x;   // 262144
  int nrow = i >> 9, k = i & 511;
  out[i] = (__bf16)in[((size_t)k << 9) + nrow];
}

// Embedding gather: x_bf[b,t,:] = emb[ti[b,t],:]; also write Z x-section
__global__ void k_embed(const int* __restrict__ ti, const float* __restrict__ emb,
                        __bf16* __restrict__ xbf, __bf16* __restrict__ Z) {
  int row = blockIdx.x;                       // b*T + t
  const float* er = emb + (size_t)ti[row] * En;
  for (int h = threadIdx.x; h < En; h += 256) {
    __bf16 e = (__bf16)er[h];
    xbf[(size_t)row * En + h]          = e;
    Z[(size_t)row * 1536 + 1024 + h]   = e;   // [h1 | ctx | x]
  }
}

// Init recurrent state (fp32 mirrors + bf16 seeds for step-0 GEMM inputs)
__global__ void k_init_state(const float* __restrict__ h0, const float* __restrict__ c0,
                             float* __restrict__ h, float* __restrict__ c,
                             __bf16* __restrict__ u0, __bf16* __restrict__ u1,
                             __bf16* __restrict__ hq) {
  int idx = blockIdx.x * 256 + threadIdx.x;   // 2*32*512 = 32768
  h[idx] = h0[idx];
  c[idx] = c0[idx];
  int l = idx >> 14, r = idx & 16383;
  int b = r >> 9, j = r & 511;
  __bf16 hb = (__bf16)h0[idx];
  if (l == 0) u0[b * 1536 + 1024 + j] = hb;                          // layer0 h_prev
  else        { u1[b * 1024 + 512 + j] = hb; hq[b * 512 + j] = hb; } // layer1 h_prev / query
}

// ---------------------------------------------------------------------------
// Shared WMMA tile compute: one wave's 32x32 of C from 64xBK LDS tiles
// ---------------------------------------------------------------------------
__device__ __forceinline__ void wmma_compute64(const __bf16 (*As)[72], const __bf16 (*Ws)[72],
                                               int wm, int wn, int r15, int kbA, int kbB,
                                               v8f acc[2][2]) {
  for (int kk = 0; kk < 64; kk += 32) {
    v16bf af[2], bfr[2];
    for (int i = 0; i < 2; ++i) {
      int row = wm + i * 16 + r15;
      af[i] = cat8(*reinterpret_cast<const v8bf*>(&As[row][kk + kbA]),
                   *reinterpret_cast<const v8bf*>(&As[row][kk + 16 + kbA]));
    }
    for (int j = 0; j < 2; ++j) {
      int col = wn + j * 16 + r15;
      bfr[j] = cat8(*reinterpret_cast<const v8bf*>(&Ws[col][kk + kbB]),
                    *reinterpret_cast<const v8bf*>(&Ws[col][kk + kbB + 16]));
    }
    for (int i = 0; i < 2; ++i)
      for (int j = 0; j < 2; ++j)
        acc[i][j] = __builtin_amdgcn_wmma_f32_16x16x32_bf16(
            false, af[i], false, bfr[j], (short)0, acc[i][j], false, false);
  }
}

// ---------------------------------------------------------------------------
// TN bf16 WMMA GEMM: C[M,N] = A[M,K] * W[N,K]^T (+bias), BM=64 BN=128 BK=64
// grid = (N/128, M/64), 256 threads (8 waves, each 32x32 of C).
// TDM path: double-buffered LDS, wave0 DMAs next tiles via tensor_load_to_lds
// while all waves run WMMA on the current buffer (TENSORcnt-synchronized).
// ---------------------------------------------------------------------------
__global__ __launch_bounds__(256) void k_gemm_tn(
    const __bf16* __restrict__ A, int lda,
    const __bf16* __restrict__ W,          // [N,K], ldw == K
    float* __restrict__ C, int ldc,
    const float* __restrict__ bias, int K) {
  const int tid = threadIdx.x, lane = tid & 31, wave = tid >> 5;
  const int n0 = blockIdx.x * 128, m0 = blockIdx.y * 64;
  const int wm = (wave >> 2) * 32;   // 0 / 32
  const int wn = (wave & 3) * 32;    // 0..96
  const int r15 = lane & 15;
  const int kbA = (lane >> 4) * 8;   // A-frag K sub-offset
  const int kbB = (lane >> 4) * 8;   // B-frag K sub-offset
  v8f acc[2][2];
  for (int i = 0; i < 2; ++i)
    for (int j = 0; j < 2; ++j) acc[i][j] = (v8f){0,0,0,0,0,0,0,0};

#if defined(USE_TDM)
  __shared__ __align__(16) __bf16 As[2][64][72];
  __shared__ __align__(16) __bf16 Ws[2][128][72];
  const unsigned ldsA0 = (unsigned)(size_t)&As[0][0][0];
  const unsigned ldsA1 = (unsigned)(size_t)&As[1][0][0];
  const unsigned ldsW0 = (unsigned)(size_t)&Ws[0][0][0];
  const unsigned ldsW1 = (unsigned)(size_t)&Ws[1][0][0];
  if (wave == 0) {                                  // prologue: DMA tiles for k0=0
    tdm_load_tile(A + (size_t)m0 * lda, ldsA0, 64, lda);
    tdm_load_tile(W + (size_t)n0 * K,  ldsW0, 128, K);
  }
  int it = 0;
  for (int k0 = 0; k0 < K; k0 += 64, ++it) {
    int cur = it & 1;
    if (k0 + 64 < K) {
      if (wave == 0) {                              // DMA next tiles into other buffer
        tdm_load_tile(A + (size_t)m0 * lda + (k0 + 64), cur ? ldsA0 : ldsA1, 64, lda);
        tdm_load_tile(W + (size_t)n0 * K  + (k0 + 64), cur ? ldsW0 : ldsW1, 128, K);
      }
      __builtin_amdgcn_s_wait_tensorcnt(2);         // current buffer's 2 DMAs done
    } else {
      __builtin_amdgcn_s_wait_tensorcnt(0);
    }
    __syncthreads();
    wmma_compute64(As[cur], Ws[cur], wm, wn, r15, kbA, kbB, acc);
    __syncthreads();                                // protect buffer before next overwrite
  }
#else
  // Fallback: single LDS buffer, register-staged global->LDS pipelining
  __shared__ __align__(16) __bf16 As[64][72];
  __shared__ __align__(16) __bf16 Ws[128][72];
  v8bf ra[2], rw[4];
  int rA[2], cA[2], rW[4], cW[4];
  for (int i = 0; i < 2; ++i) {                     // A tile 64x64 = 512 segs
    int seg = tid + i * 256; rA[i] = seg >> 3; cA[i] = (seg & 7) * 8;
    ra[i] = *reinterpret_cast<const v8bf*>(A + (size_t)(m0 + rA[i]) * lda + cA[i]);
  }
  for (int i = 0; i < 4; ++i) {                     // W tile 128x64 = 1024 segs
    int seg = tid + i * 256; rW[i] = seg >> 3; cW[i] = (seg & 7) * 8;
    rw[i] = *reinterpret_cast<const v8bf*>(W + (size_t)(n0 + rW[i]) * K + cW[i]);
  }
  for (int k0 = 0; k0 < K; k0 += 64) {
    for (int i = 0; i < 2; ++i) *reinterpret_cast<v8bf*>(&As[rA[i]][cA[i]]) = ra[i];
    for (int i = 0; i < 4; ++i) *reinterpret_cast<v8bf*>(&Ws[rW[i]][cW[i]]) = rw[i];
    __syncthreads();
    if (k0 + 64 < K) {                              // issue next tile's loads early
      for (int i = 0; i < 2; ++i)
        ra[i] = *reinterpret_cast<const v8bf*>(A + (size_t)(m0 + rA[i]) * lda + k0 + 64 + cA[i]);
      for (int i = 0; i < 4; ++i)
        rw[i] = *reinterpret_cast<const v8bf*>(W + (size_t)(n0 + rW[i]) * K + k0 + 64 + cW[i]);
    }
    wmma_compute64(As, Ws, wm, wn, r15, kbA, kbB, acc);
    __syncthreads();
  }
#endif

  const int mrow = (lane >> 4) * 8;
  for (int i = 0; i < 2; ++i)
    for (int j = 0; j < 2; ++j) {
      int n = n0 + wn + j * 16 + r15;
      float bv = bias ? bias[n] : 0.f;
      float* cp = C + (size_t)(m0 + wm + i * 16 + mrow) * ldc + n;
      for (int r = 0; r < 8; ++r) cp[(size_t)r * ldc] = acc[i][j][r] + bv;
    }
}

// ---------------------------------------------------------------------------
// Skinny M=32 TN bf16 WMMA GEMM: C[32,N] = A[32,K] * W[N,K]^T
// grid = N/128, 256 threads; wave w owns n-tile w, both m-tiles.
// Register-staged pipelining (runs 192x inside the sequential scan).
// ---------------------------------------------------------------------------
__global__ __launch_bounds__(256) void k_gemm32(
    const __bf16* __restrict__ A, const __bf16* __restrict__ W,
    float* __restrict__ C, int N, int K) {
  __shared__ __align__(16) __bf16 As[32][72];
  __shared__ __align__(16) __bf16 Ws[128][72];
  const int tid = threadIdx.x, lane = tid & 31, wave = tid >> 5;
  const int n0 = blockIdx.x * 128;
  const int r15 = lane & 15;
  const int kbA = (lane >> 4) * 8, kbB = (lane >> 4) * 8;
  v8f acc0 = (v8f){0,0,0,0,0,0,0,0}, acc1 = (v8f){0,0,0,0,0,0,0,0};

  const int rA = tid >> 3, cA = (tid & 7) * 8;      // A tile 32x64 = 256 segs
  v8bf ra = *reinterpret_cast<const v8bf*>(A + (size_t)rA * K + cA);
  v8bf rw[4];
  int rW[4], cW[4];
  for (int i = 0; i < 4; ++i) {
    int seg = tid + i * 256; rW[i] = seg >> 3; cW[i] = (seg & 7) * 8;
    rw[i] = *reinterpret_cast<const v8bf*>(W + (size_t)(n0 + rW[i]) * K + cW[i]);
  }

  for (int k0 = 0; k0 < K; k0 += 64) {
    *reinterpret_cast<v8bf*>(&As[rA][cA]) = ra;
    for (int i = 0; i < 4; ++i) *reinterpret_cast<v8bf*>(&Ws[rW[i]][cW[i]]) = rw[i];
    __syncthreads();
    if (k0 + 64 < K) {
      ra = *reinterpret_cast<const v8bf*>(A + (size_t)rA * K + k0 + 64 + cA);
      for (int i = 0; i < 4; ++i)
        rw[i] = *reinterpret_cast<const v8bf*>(W + (size_t)(n0 + rW[i]) * K + k0 + 64 + cW[i]);
    }
    for (int kk = 0; kk < 64; kk += 32) {
      int col = wave * 16 + r15;
      v16bf bfr = cat8(*reinterpret_cast<const v8bf*>(&Ws[col][kk + kbB]),
                       *reinterpret_cast<const v8bf*>(&Ws[col][kk + kbB + 16]));
      v16bf a0 = cat8(*reinterpret_cast<const v8bf*>(&As[r15][kk + kbA]),
                      *reinterpret_cast<const v8bf*>(&As[r15][kk + 16 + kbA]));
      v16bf a1 = cat8(*reinterpret_cast<const v8bf*>(&As[16 + r15][kk + kbA]),
                      *reinterpret_cast<const v8bf*>(&As[16 + r15][kk + 16 + kbA]));
      acc0 = __builtin_amdgcn_wmma_f32_16x16x32_bf16(false, a0, false, bfr, (short)0, acc0, false, false);
      acc1 = __builtin_amdgcn_wmma_f32_16x16x32_bf16(false, a1, false, bfr, (short)0, acc1, false, false);
    }
    __syncthreads();
  }
  const int mrow = (lane >> 4) * 8;
  int n = n0 + wave * 16 + r15;
  for (int r = 0; r < 8; ++r) {
    C[(size_t)(mrow + r) * N + n]      = acc0[r];
    C[(size_t)(16 + mrow + r) * N + n] = acc1[r];
  }
}

// ---------------------------------------------------------------------------
// Attention: scores = v_a . tanh(enc_proj + qW) + mask; softmax; ctx.
// One block per batch row. Also copies x_t into u0, ctx into u0 & Z.
// ---------------------------------------------------------------------------
__global__ void k_attn(const float* __restrict__ encprj, const float* __restrict__ qW,
                       const float* __restrict__ enc, const float* __restrict__ va,
                       const unsigned char* __restrict__ smask,
                       const __bf16* __restrict__ xbf, __bf16* __restrict__ u0,
                       __bf16* __restrict__ Z, float* __restrict__ attn_out, int t) {
  const int b = blockIdx.x, tid = threadIdx.x;
  __shared__ float part[256];
  __shared__ float sc[Sn];
  __shared__ float w[Sn];
  __shared__ float red[2];
  int s = tid & 63, q = tid >> 6;
  const float* ep = encprj + ((size_t)b * Sn + s) * Hn;
  const float* qp = qW + (size_t)b * Hn;
  float p = 0.f;
  for (int h = q * 128; h < q * 128 + 128; ++h) p += va[h] * tanhf(ep[h] + qp[h]);
  part[tid] = p;
  __syncthreads();
  if (tid < Sn) {
    float v = part[tid] + part[tid + 64] + part[tid + 128] + part[tid + 192];
    if (smask[b * Sn + tid]) v -= 1e9f;
    sc[tid] = v;
  }
  __syncthreads();
  if (tid == 0) {
    float mx = sc[0];
    for (int i = 1; i < Sn; ++i) mx = fmaxf(mx, sc[i]);
    red[0] = mx;
  }
  __syncthreads();
  if (tid < Sn) w[tid] = __expf(sc[tid] - red[0]);
  __syncthreads();
  if (tid == 0) {
    float sm = 0.f;
    for (int i = 0; i < Sn; ++i) sm += w[i];
    red[1] = 1.f / sm;
  }
  __syncthreads();
  size_t row = (size_t)b * Tn + t;
  if (tid < Sn) {
    float wn = w[tid] * red[1];
    w[tid] = wn;
    attn_out[row * Sn + tid] = wn;
  }
  __syncthreads();
  for (int h = tid; h < Hn; h += 256) {
    float cv = 0.f;
    const float* eb = enc + (size_t)b * Sn * Hn + h;
    for (int s2 = 0; s2 < Sn; ++s2) cv += w[s2] * eb[(size_t)s2 * Hn];
    __bf16 cb = (__bf16)cv;
    u0[b * 1536 + 512 + h]    = cb;
    Z[row * 1536 + 512 + h]   = cb;
  }
  for (int h = tid; h < En; h += 256) u0[b * 1536 + h] = xbf[row * En + h];
}

// ---------------------------------------------------------------------------
// LSTM cells (gate order i,f,g,o along the 4H axis)
// ---------------------------------------------------------------------------
__global__ void k_cell0(const float* __restrict__ g, const float* __restrict__ bih,
                        const float* __restrict__ bhh, float* __restrict__ hst,
                        float* __restrict__ cst, __bf16* __restrict__ u1,
                        __bf16* __restrict__ u0) {
  int idx = blockIdx.x * 256 + threadIdx.x;  // 32*512
  int b = idx >> 9, n = idx & 511;
  const float* gr = g + (size_t)b * 2048;
  float gi = gr[n]        + bih[n]        + bhh[n];
  float gf = gr[512 + n]  + bih[512 + n]  + bhh[512 + n];
  float gg = gr[1024 + n] + bih[1024 + n] + bhh[1024 + n];
  float go = gr[1536 + n] + bih[1536 + n] + bhh[1536 + n];
  float c2 = sigf(gf) * cst[idx] + sigf(gi) * tanhf(gg);
  float h2 = sigf(go) * tanhf(c2);
  cst[idx] = c2; hst[idx] = h2;
  __bf16 hb = (__bf16)h2;
  u1[b * 1024 + n]        = hb;   // layer1 input
  u0[b * 1536 + 1024 + n] = hb;   // next step's layer0 h_prev
}

__global__ void k_cell1(const float* __restrict__ g, const float* __restrict__ bih,
                        const float* __restrict__ bhh, float* __restrict__ hst,
                        float* __restrict__ cst, __bf16* __restrict__ u1,
                        __bf16* __restrict__ hq, __bf16* __restrict__ Z, int t) {
  int idx = blockIdx.x * 256 + threadIdx.x;
  int b = idx >> 9, n = idx & 511;
  const float* gr = g + (size_t)b * 2048;
  float gi = gr[n]        + bih[n]        + bhh[n];
  float gf = gr[512 + n]  + bih[512 + n]  + bhh[512 + n];
  float gg = gr[1024 + n] + bih[1024 + n] + bhh[1024 + n];
  float go = gr[1536 + n] + bih[1536 + n] + bhh[1536 + n];
  float c2 = sigf(gf) * cst[idx] + sigf(gi) * tanhf(gg);
  float h2 = sigf(go) * tanhf(c2);
  cst[idx] = c2; hst[idx] = h2;
  __bf16 hb = (__bf16)h2;
  u1[b * 1024 + 512 + n] = hb;                          // next step's layer1 h_prev
  hq[b * 512 + n]        = hb;                          // next step's attention query
  Z[((size_t)b * Tn + t) * 1536 + n] = hb;              // logits input, h1 section
}

__global__ void k_finalize(const float* __restrict__ h, const float* __restrict__ c,
                           float* __restrict__ oh, float* __restrict__ oc) {
  int idx = blockIdx.x * 256 + threadIdx.x;  // 32768
  oh[idx] = h[idx];
  oc[idx] = c[idx];
}

// ---------------------------------------------------------------------------
// Host orchestration
// ---------------------------------------------------------------------------
extern "C" void kernel_launch(void* const* d_in, const int* in_sizes, int n_in,
                              void* d_out, int out_size, void* d_ws, size_t ws_size,
                              hipStream_t stream) {
  (void)in_sizes; (void)n_in; (void)out_size; (void)ws_size;
  const int*   ti     = (const int*)d_in[0];
  const float* enc    = (const float*)d_in[2];
  const float* h0     = (const float*)d_in[3];
  const float* c0     = (const float*)d_in[4];
  const unsigned char* smask = (const unsigned char*)d_in[5];
  const float* emb    = (const float*)d_in[6];
  const float* WaEnc  = (const float*)d_in[7];
  const float* WaDec  = (const float*)d_in[8];
  const float* va     = (const float*)d_in[9];
  const float* Wih0   = (const float*)d_in[10];
  const float* Whh0   = (const float*)d_in[11];
  const float* bih0   = (const float*)d_in[12];
  const float* bhh0   = (const float*)d_in[13];
  const float* Wih1   = (const float*)d_in[14];
  const float* Whh1   = (const float*)d_in[15];
  const float* bih1   = (const float*)d_in[16];
  const float* bhh1   = (const float*)d_in[17];
  const float* Wout   = (const float*)d_in[18];
  const float* bout   = (const float*)d_in[19];

  char* ws = (char*)d_ws;
  // workspace layout (bytes)
  __bf16* WoutBf  = (__bf16*)(ws + 0);                    //  98,304,000
  __bf16* Wcat0   = (__bf16*)(ws + 98304000);             //   6,291,456
  __bf16* Wcat1   = (__bf16*)(ws + 104595456);            //   4,194,304
  __bf16* WaEncT  = (__bf16*)(ws + 108789760);            //     524,288
  __bf16* WaDecT  = (__bf16*)(ws + 109314048);            //     524,288
  __bf16* encBf   = (__bf16*)(ws + 109838336);            //   2,097,152
  __bf16* xBf     = (__bf16*)(ws + 111935488);            //   2,097,152
  __bf16* Zbf     = (__bf16*)(ws + 114032640);            //   6,291,456
  float*  encPrj  = (float*) (ws + 120324096);            //   4,194,304
  float*  qW      = (float*) (ws + 124518400);            //      65,536
  float*  g0      = (float*) (ws + 124583936);            //     262,144
  float*  g1      = (float*) (ws + 124846080);            //     262,144
  __bf16* u0      = (__bf16*)(ws + 125108224);            //      98,304
  __bf16* u1      = (__bf16*)(ws + 125206528);            //      65,536
  __bf16* hq      = (__bf16*)(ws + 125272064);            //      32,768
  float*  hSt     = (float*) (ws + 125304832);            //     131,072
  float*  cSt     = (float*) (ws + 125435904);            //     131,072

  float* outLogits = (float*)d_out;                       // [B,T,V]
  float* outH      = outLogits + (size_t)Bn * Tn * Vn;    // [L,B,H]
  float* outC      = outH + 2 * Bn * Hn;
  float* outAttn   = outC + 2 * Bn * Hn;                  // [B,T,S]

  // ---- one-time precompute (per launch; deterministic) ----
  k_cvt_bf16<<<(Bn*Sn*Hn + 255)/256, 256, 0, stream>>>(enc, encBf, Bn*Sn*Hn);
  k_cvt_bf16<<<((int)((size_t)Vn*1536) + 255)/256, 256, 0, stream>>>(Wout, WoutBf, Vn*1536);
  k_cat_w<<<(2048*1536 + 255)/256, 256, 0, stream>>>(Wih0, Whh0, Wcat0, 1024, 512);
  k_cat_w<<<(2048*1024 + 255)/256, 256, 0, stream>>>(Wih1, Whh1, Wcat1, 512, 512);
  k_transpose512<<<1024, 256, 0, stream>>>(WaEnc, WaEncT);
  k_transpose512<<<1024, 256, 0, stream>>>(WaDec, WaDecT);
  k_embed<<<Bn*Tn, 256, 0, stream>>>(ti, emb, xBf, Zbf);
  k_init_state<<<128, 256, 0, stream>>>(h0, c0, hSt, cSt, u0, u1, hq);
  // enc_proj = enc @ Wa_enc  (hoisted, loop-invariant)
  k_gemm_tn<<<dim3(Hn/128, (Bn*Sn)/64), 256, 0, stream>>>(encBf, Hn, WaEncT, encPrj, Hn, nullptr, Hn);

  // ---- sequential scan over T ----
  for (int t = 0; t < Tn; ++t) {
    k_gemm32<<<Hn/128, 256, 0, stream>>>(hq, WaDecT, qW, Hn, Hn);              // q @ Wa_dec
    k_attn<<<Bn, 256, 0, stream>>>(encPrj, qW, enc, va, smask, xBf, u0, Zbf, outAttn, t);
    k_gemm32<<<2048/128, 256, 0, stream>>>(u0, Wcat0, g0, 2048, 1536);         // layer0 gates
    k_cell0<<<(Bn*Hn)/256, 256, 0, stream>>>(g0, bih0, bhh0, hSt, cSt, u1, u0);
    k_gemm32<<<2048/128, 256, 0, stream>>>(u1, Wcat1, g1, 2048, 1024);         // layer1 gates
    k_cell1<<<(Bn*Hn)/256, 256, 0, stream>>>(g1, bih1, bhh1, hSt + Bn*Hn, cSt + Bn*Hn,
                                             u1, hq, Zbf, t);
  }

  // ---- batched vocab projection: logits = Z @ W_out^T + b_out ----
  k_gemm_tn<<<dim3(Vn/128, (Bn*Tn)/64), 256, 0, stream>>>(Zbf, 1536, WoutBf,
                                                          outLogits, Vn, bout, 1536);
  k_finalize<<<128, 256, 0, stream>>>(hSt, cSt, outH, outC);
}
